// FullAttention_Spatial_11269994184769
// MI455X (gfx1250) — compile-verified, matrix operand order checked by reading, not run on hardware
//
#include <hip/hip_runtime.h>

// Problem constants (from reference)
#define N_ 4
#define L_ 1024
#define S_ 1024
#define H_ 16
#define E_ 64
#define D_ 64

typedef __attribute__((ext_vector_type(16))) __bf16 v16bf;
typedef __attribute__((ext_vector_type(8)))  __bf16 bf16x8;
typedef __attribute__((ext_vector_type(8)))  float  v8f;

#define LOG2E 1.4426950408889634f

__device__ __forceinline__ v16bf make_frag(bf16x8 lo, bf16x8 hi) {
  v16bf f;
#pragma unroll
  for (int i = 0; i < 8; ++i) { f[i] = lo[i]; f[i + 8] = hi[i]; }
  return f;
}

__global__ __launch_bounds__(256)
void attn_graphbias_kernel(const float* __restrict__ Q, const float* __restrict__ K,
                           const float* __restrict__ V, const float* __restrict__ AM,
                           const float* __restrict__ KLM, const float* __restrict__ PT,
                           const int* __restrict__ POS, float* __restrict__ OUT)
{
  // LDS tiles: K chunk row-major bf16, V chunk transposed bf16, per-wave prob scratch
  __shared__ __align__(16) __bf16 Ksh[32][72];      // [s][e], padded
  __shared__ __align__(16) __bf16 Vt[64][40];       // [d][s], padded
  __shared__ __align__(16) __bf16 Psh[8][16][40];   // per-wave [m][s]

  const int nh   = blockIdx.x;
  const int n    = nh / H_;
  const int h    = nh % H_;
  const int tid  = threadIdx.x;
  const int wv   = tid >> 5;
  const int lane = tid & 31;
  const int half = lane >> 4;      // which 16-lane half
  const int nidx = lane & 15;      // row (A) / column (B,C,D) index within tile
  const int b0   = half * 8;       // k-chunk base per WMMA 16-bit layout
  const int lbase = blockIdx.y * 128 + wv * 16;

  // log2-domain softmax: fold LOG2E into the scale and the pos-bias table
  const float scl = 0.125f * LOG2E;            // (1/sqrt(E)) * log2(e)
  const float pt0 = PT[h]      * LOG2E;        // pos_table[0][h] * log2(e)
  const float dpt = PT[H_ + h] * LOG2E - pt0;  // (pt1 - pt0) in log2 domain

  // ---- Q A-fragments (16 rows x E=64 -> two 16x32 bf16 fragments), loaded once ----
  const float* qrow = Q + (((size_t)n * L_ + lbase + nidx) * H_ + h) * E_;
  v16bf qa[2];
#pragma unroll
  for (int ks = 0; ks < 2; ++ks) {
    v16bf f;
#pragma unroll
    for (int i = 0; i < 8; ++i) {
      f[i]     = (__bf16)qrow[ks * 32 + b0 + i];
      f[i + 8] = (__bf16)qrow[ks * 32 + b0 + 16 + i];
    }
    qa[ks] = f;
  }

  // all-ones B fragment: one extra WMMA per chunk computes the row sums
  v16bf onesb;
#pragma unroll
  for (int i = 0; i < 16; ++i) onesb[i] = (__bf16)1.0f;

  const v8f vzero = {};
  v8f oacc[4];
#pragma unroll
  for (int t = 0; t < 4; ++t) oacc[t] = vzero;
  v8f sumacc = vzero;                          // row-sum accumulator (P·1)
  float rmax[8];
#pragma unroll
  for (int j = 0; j < 8; ++j) rmax[j] = -3.0e38f;

  // cooperative K/V -> LDS indices: 256 threads cover 32 s-rows x 64 cols, 8 cols each
  const int cs = tid >> 3;         // s within chunk (0..31)
  const int ce = (tid & 7) * 8;    // e/d base (0..56)
  const float* kbase = K + (((size_t)n * S_ + cs) * H_ + h) * E_ + ce;
  const float* vbase = V + (((size_t)n * S_ + cs) * H_ + h) * D_ + ce;
  const size_t srow = (size_t)H_ * E_;            // floats per s step (=1024)
  const size_t schunk = 32 * srow;

  // ---- software pipeline: prefetch chunk 0 into registers ----
  float kreg[8], vreg[8];
#pragma unroll
  for (int i = 0; i < 8; ++i) { kreg[i] = kbase[i]; vreg[i] = vbase[i]; }

  for (int sb = 0; sb < S_; sb += 32) {
    __syncthreads();                               // previous chunk's LDS reads done
    // publish staged chunk to LDS (bf16), V transposed
#pragma unroll
    for (int i = 0; i < 8; ++i) Ksh[cs][ce + i] = (__bf16)kreg[i];
#pragma unroll
    for (int i = 0; i < 8; ++i) Vt[ce + i][cs] = (__bf16)vreg[i];
    __syncthreads();                               // chunk visible to all waves

    // issue next chunk's global loads now; latency hides under compute below
    if (sb + 32 < S_) {
      const float* kp = kbase + (size_t)(sb + 32) * srow;
      const float* vp = vbase + (size_t)(sb + 32) * srow;
#pragma unroll
      for (int i = 0; i < 8; ++i) { kreg[i] = kp[i]; vreg[i] = vp[i]; }
      if (sb + 64 < S_) {                          // L2 prefetch two chunks ahead
        __builtin_prefetch(kp + schunk, 0, 1);
        __builtin_prefetch(vp + schunk, 0, 1);
      }
    }

    // ---- QK^T: load ALL four K B-fragments as one DS clause ----
    v16bf kb00, kb01, kb10, kb11;                  // kb[kstep][tile]
    {
      bf16x8 l00 = *(const bf16x8*)&Ksh[nidx][b0];
      bf16x8 h00 = *(const bf16x8*)&Ksh[nidx][b0 + 16];
      bf16x8 l01 = *(const bf16x8*)&Ksh[16 + nidx][b0];
      bf16x8 h01 = *(const bf16x8*)&Ksh[16 + nidx][b0 + 16];
      bf16x8 l10 = *(const bf16x8*)&Ksh[nidx][32 + b0];
      bf16x8 h10 = *(const bf16x8*)&Ksh[nidx][32 + b0 + 16];
      bf16x8 l11 = *(const bf16x8*)&Ksh[16 + nidx][32 + b0];
      bf16x8 h11 = *(const bf16x8*)&Ksh[16 + nidx][32 + b0 + 16];
      kb00 = make_frag(l00, h00);
      kb01 = make_frag(l01, h01);
      kb10 = make_frag(l10, h10);
      kb11 = make_frag(l11, h11);
    }
    // pin schedule: all ds_loads above must issue before the WMMA chain below,
    // so the four fragments stay in distinct registers (one DS wait, not four)
    asm volatile("" ::: "memory");

    v8f sc0 = vzero, sc1 = vzero;
    sc0 = __builtin_amdgcn_wmma_f32_16x16x32_bf16(false, qa[0], false, kb00,
                                                  (short)0, sc0, false, false);
    sc1 = __builtin_amdgcn_wmma_f32_16x16x32_bf16(false, qa[0], false, kb01,
                                                  (short)0, sc1, false, false);
    sc0 = __builtin_amdgcn_wmma_f32_16x16x32_bf16(false, qa[1], false, kb10,
                                                  (short)0, sc0, false, false);
    sc1 = __builtin_amdgcn_wmma_f32_16x16x32_bf16(false, qa[1], false, kb11,
                                                  (short)0, sc1, false, false);

    // ---- masks + graph positional bias (branch-free fma select, log2 domain) ----
    const int s0 = sb + nidx;
    const int s1 = sb + 16 + nidx;
    const float klm0 = KLM[n * S_ + s0];
    const float klm1 = KLM[n * S_ + s1];
#pragma unroll
    for (int j = 0; j < 8; ++j) {
      const int l = lbase + half * 8 + j;
      const float a0 = AM[(size_t)l * S_ + s0];
      const float a1 = AM[(size_t)l * S_ + s1];
      const float p0f = (float)POS[((size_t)n * L_ + l) * S_ + s0];
      const float p1f = (float)POS[((size_t)n * L_ + l) * S_ + s1];
      sc0[j] = fmaf(scl, sc0[j] + a0 + klm0, fmaf(p0f, dpt, pt0));
      sc1[j] = fmaf(scl, sc1[j] + a1 + klm1, fmaf(p1f, dpt, pt0));
    }

    // ---- online softmax: only the max needs a cross-lane reduction ----
#pragma unroll
    for (int j = 0; j < 8; ++j) {
      float cm = fmaxf(sc0[j], sc1[j]);
#pragma unroll
      for (int off = 8; off >= 1; off >>= 1)
        cm = fmaxf(cm, __shfl_xor(cm, off, 32));
      const float nm = fmaxf(rmax[j], cm);
      const float alpha = exp2f(rmax[j] - nm);
      rmax[j] = nm;
      sc0[j] = exp2f(sc0[j] - nm);
      sc1[j] = exp2f(sc1[j] - nm);
      sumacc[j] *= alpha;                          // row-sum follows same recurrence
#pragma unroll
      for (int t = 0; t < 4; ++t) oacc[t][j] *= alpha;
    }

    // ---- D-layout -> A-layout transpose of probs via per-wave LDS (same-wave DS in-order) ----
#pragma unroll
    for (int j = 0; j < 8; ++j) {
      Psh[wv][half * 8 + j][nidx]      = (__bf16)sc0[j];
      Psh[wv][half * 8 + j][16 + nidx] = (__bf16)sc1[j];
    }
    asm volatile("" ::: "memory");

    // ---- load prob A-fragment + ALL four V B-fragments as one DS clause ----
    bf16x8 plo = *(const bf16x8*)&Psh[wv][nidx][b0];
    bf16x8 phi = *(const bf16x8*)&Psh[wv][nidx][b0 + 16];
    v16bf vb0, vb1, vb2, vb3;
    {
      bf16x8 l0 = *(const bf16x8*)&Vt[nidx][b0];
      bf16x8 h0 = *(const bf16x8*)&Vt[nidx][b0 + 16];
      bf16x8 l1 = *(const bf16x8*)&Vt[16 + nidx][b0];
      bf16x8 h1 = *(const bf16x8*)&Vt[16 + nidx][b0 + 16];
      bf16x8 l2 = *(const bf16x8*)&Vt[32 + nidx][b0];
      bf16x8 h2 = *(const bf16x8*)&Vt[32 + nidx][b0 + 16];
      bf16x8 l3 = *(const bf16x8*)&Vt[48 + nidx][b0];
      bf16x8 h3 = *(const bf16x8*)&Vt[48 + nidx][b0 + 16];
      vb0 = make_frag(l0, h0);
      vb1 = make_frag(l1, h1);
      vb2 = make_frag(l2, h2);
      vb3 = make_frag(l3, h3);
    }
    v16bf pa = make_frag(plo, phi);
    asm volatile("" ::: "memory");                 // batch DS waits before WMMA burst

    sumacc  = __builtin_amdgcn_wmma_f32_16x16x32_bf16(false, pa, false, onesb,
                                                      (short)0, sumacc, false, false);
    oacc[0] = __builtin_amdgcn_wmma_f32_16x16x32_bf16(false, pa, false, vb0,
                                                      (short)0, oacc[0], false, false);
    oacc[1] = __builtin_amdgcn_wmma_f32_16x16x32_bf16(false, pa, false, vb1,
                                                      (short)0, oacc[1], false, false);
    oacc[2] = __builtin_amdgcn_wmma_f32_16x16x32_bf16(false, pa, false, vb2,
                                                      (short)0, oacc[2], false, false);
    oacc[3] = __builtin_amdgcn_wmma_f32_16x16x32_bf16(false, pa, false, vb3,
                                                      (short)0, oacc[3], false, false);
  }

  // ---- epilogue: normalize and store [N,L,H,D] f32 ----
#pragma unroll
  for (int j = 0; j < 8; ++j) {
    const int l = lbase + half * 8 + j;
    const float inv = 1.0f / sumacc[j];            // every column holds the row sum
    float* orow = OUT + (((size_t)n * L_ + l) * H_ + h) * D_;
#pragma unroll
    for (int t = 0; t < 4; ++t)
      orow[t * 16 + nidx] = oacc[t][j] * inv;
  }
}

extern "C" void kernel_launch(void* const* d_in, const int* in_sizes, int n_in,
                              void* d_out, int out_size, void* d_ws, size_t ws_size,
                              hipStream_t stream) {
  (void)in_sizes; (void)n_in; (void)out_size; (void)d_ws; (void)ws_size;
  const float* Q   = (const float*)d_in[0];
  const float* K   = (const float*)d_in[1];
  const float* V   = (const float*)d_in[2];
  const float* AM  = (const float*)d_in[3];
  const float* KLM = (const float*)d_in[4];
  const float* PT  = (const float*)d_in[5];
  const int*   POS = (const int*)d_in[6];
  float* OUT = (float*)d_out;

  dim3 grid(N_ * H_, L_ / 128);   // 64 x 8 blocks
  dim3 block(256);                // 8 waves (wave32)
  hipLaunchKernelGGL(attn_graphbias_kernel, grid, block, 0, stream,
                     Q, K, V, AM, KLM, PT, POS, OUT);
}
